// SAGE_66718021976360
// MI455X (gfx1250) — matrix operand order repaired and checked by reading, not compiled, not run on hardware
//
#include <hip/hip_runtime.h>

// ---------- types for WMMA ----------
typedef __attribute__((ext_vector_type(16))) __bf16 v16bf;
typedef __attribute__((ext_vector_type(8)))  float  v8f;

// Native f32 -> bf16 convert: clang lowers fptrunc to the gfx1250 hardware
// cvt (packed v_cvt_pk_bf16_f32 when it can fuse pairs), replacing the old
// 3-4 VALU/element integer RNE emulation.
__device__ __forceinline__ __bf16 f2bf(float f) { return (__bf16)f; }

// ---------- constants from the reference ----------
constexpr int NN = 100000;    // nodes
constexpr int NE = 1600000;   // edges
constexpr int NG = 1000;      // graphs

// ---------- zero workspace ----------
__global__ void zero_kernel(float* p, size_t n) {
    size_t i = (size_t)blockIdx.x * blockDim.x + threadIdx.x;
    size_t stride = (size_t)gridDim.x * blockDim.x;
    for (; i < n; i += stride) p[i] = 0.0f;
}

// ---------- edge scatter: one wave32 per edge, lane = feature ----------
__global__ void scatter32_kernel(const float* __restrict__ x,
                                 const int* __restrict__ ei,
                                 float* __restrict__ agg, int nE) {
    int wave = (int)((blockIdx.x * (size_t)blockDim.x + threadIdx.x) >> 5);
    int lane = threadIdx.x & 31;
    if (wave >= nE) return;
    int src = ei[wave];
    int dst = ei[nE + wave];
    atomicAdd(&agg[(size_t)dst * 32 + lane], x[(size_t)src * 32 + lane]);
}

__global__ void scatter64_kernel(const float* __restrict__ h,
                                 const int* __restrict__ ei,
                                 float* __restrict__ agg, int nE) {
    int wave = (int)((blockIdx.x * (size_t)blockDim.x + threadIdx.x) >> 5);
    int lane = threadIdx.x & 31;
    if (wave >= nE) return;
    int src = ei[wave];
    int dst = ei[nE + wave];
    atomicAdd(&agg[(size_t)dst * 64 + lane],      h[(size_t)src * 64 + lane]);
    atomicAdd(&agg[(size_t)dst * 64 + 32 + lane], h[(size_t)src * 64 + 32 + lane]);
}

// Fill LDS with B-matrix fragments in native per-lane WMMA layout:
// frag f = ks*ntiles + nt ; element (lane, e) holds W[K][N] with
// K = ks*32 + (e&7) + 8*(lane>>4) + 16*(e>>3), N = nt*16 + (lane&15).
// Stored linearly so each lane's 16 bf16 elements are 32B contiguous.
__device__ __forceinline__ void fill_wfrag(v16bf* frag, const float* __restrict__ W,
                                           int Kdim, int Ndim, int tid, int nthreads) {
    int ntiles = Ndim >> 4;
    int total = (Kdim >> 5) * ntiles * 512;          // frags * 32 lanes * 16 elems
    __bf16* p = (__bf16*)frag;
    for (int idx = tid; idx < total; idx += nthreads) {
        int f   = idx >> 9;
        int rem = idx & 511;
        int l = rem >> 4, e = rem & 15;
        int ks = f / ntiles, nt = f % ntiles;
        int hh = l >> 4, n = l & 15;
        int Kg = ks * 32 + (e & 7) + 8 * hh + 16 * (e >> 3);
        int N  = nt * 16 + n;
        p[idx] = f2bf(W[(size_t)Kg * Ndim + N]);
    }
}

// Inverse of the A-fragment K mapping: given within-slab k (0..31) return
// owner half h and element e such that a_kidx(e,h)==k.
__device__ __forceinline__ void a_inv(int k, int& h, int& e) {
    h = (k >> 3) & 1;                    // [0,8)+[16,24)->h0 ; [8,16)+[24,32)->h1
    e = (k < 16) ? (k - 8 * h) : (k - 8 - 8 * h);
}

// ---------- SAGE layer GEMM: out = relu(Aagg@Wl + bias + Aself@Wr), N=64 ----------
template <int K>
__global__ void sage_gemm_kernel(const float* __restrict__ Aagg,
                                 const float* __restrict__ Aself,
                                 const float* __restrict__ Wl,
                                 const float* __restrict__ Wr,
                                 const float* __restrict__ bias,
                                 float* __restrict__ Out, int M) {
    constexpr int KS = K / 32;
    constexpr int FRAG = KS * 4;                       // fragments per weight matrix
    __shared__ v16bf sWfrag[2 * FRAG * 32];            // [which][ks][nt][lane]
    __shared__ float sB[64];

    fill_wfrag(sWfrag,             Wl, K, 64, threadIdx.x, blockDim.x);
    fill_wfrag(sWfrag + FRAG * 32, Wr, K, 64, threadIdx.x, blockDim.x);
    if (threadIdx.x < 64) sB[threadIdx.x] = bias[threadIdx.x];
    __syncthreads();

    int lane = threadIdx.x & 31;
    int tile = blockIdx.x * (blockDim.x >> 5) + (threadIdx.x >> 5);
    int rowbase = tile * 16;
    if (rowbase >= M) return;
    int half = lane >> 4;
    int m = lane & 15;   // A row within tile
    int n = lane & 15;   // C/D column

    v8f acc[4];
    #pragma unroll
    for (int t = 0; t < 4; ++t) acc[t] = (v8f){};

    #pragma unroll
    for (int which = 0; which < 2; ++which) {
        const float* A = which ? Aself : Aagg;
        const v16bf* Wf = sWfrag + which * FRAG * 32;
        #pragma unroll
        for (int ks = 0; ks < KS; ++ks) {
            const float* row = A + (size_t)(rowbase + m) * K + ks * 32;
            __builtin_prefetch(row + 16 * K, 0, 0);    // global_prefetch_b8 ahead
            int k0 = half * 8;
            float4 f0 = *(const float4*)(row + k0);
            float4 f1 = *(const float4*)(row + k0 + 4);
            float4 f2 = *(const float4*)(row + 16 + k0);
            float4 f3 = *(const float4*)(row + 16 + k0 + 4);
            v16bf a;
            a[0]=f2bf(f0.x);  a[1]=f2bf(f0.y);  a[2]=f2bf(f0.z);  a[3]=f2bf(f0.w);
            a[4]=f2bf(f1.x);  a[5]=f2bf(f1.y);  a[6]=f2bf(f1.z);  a[7]=f2bf(f1.w);
            a[8]=f2bf(f2.x);  a[9]=f2bf(f2.y);  a[10]=f2bf(f2.z); a[11]=f2bf(f2.w);
            a[12]=f2bf(f3.x); a[13]=f2bf(f3.y); a[14]=f2bf(f3.z); a[15]=f2bf(f3.w);
            #pragma unroll
            for (int nt = 0; nt < 4; ++nt) {
                v16bf b = Wf[(ks * 4 + nt) * 32 + lane];   // single 32B LDS read
                acc[nt] = __builtin_amdgcn_wmma_f32_16x16x32_bf16(
                    false, a, false, b, (short)0, acc[nt], false, false);
            }
        }
    }

    #pragma unroll
    for (int nt = 0; nt < 4; ++nt) {
        float bv = sB[nt * 16 + n];
        #pragma unroll
        for (int r = 0; r < 8; ++r) {
            float v = fmaxf(acc[nt][r] + bv, 0.0f);
            Out[(size_t)(rowbase + r + 8 * half) * 64 + nt * 16 + n] = v;
        }
    }
}

// ---------- global add pool: one wave per node ----------
__global__ void pool_kernel(const float* __restrict__ h,
                            const int* __restrict__ batch,
                            float* __restrict__ g, int nNodes) {
    int wave = (int)((blockIdx.x * (size_t)blockDim.x + threadIdx.x) >> 5);
    int lane = threadIdx.x & 31;
    if (wave >= nNodes) return;
    int b = batch[wave];
    atomicAdd(&g[(size_t)b * 64 + lane],      h[(size_t)wave * 64 + lane]);
    atomicAdd(&g[(size_t)b * 64 + 32 + lane], h[(size_t)wave * 64 + 32 + lane]);
}

// ---------- fused MLP tail: 64 -> 128 -> 64 -> 16, wave0 computes 16 graph rows ----------
__global__ void mlp_kernel(const float* __restrict__ g,
                           const float* __restrict__ wl1, const float* __restrict__ bl1,
                           const float* __restrict__ wl2, const float* __restrict__ bl2,
                           const float* __restrict__ wl3, const float* __restrict__ bl3,
                           float* __restrict__ out, int G) {
    __shared__ v16bf sW1[16 * 32];   // 2 ks * 8 nt
    __shared__ v16bf sW2[16 * 32];   // 4 ks * 4 nt
    __shared__ v16bf sW3[2 * 32];    // 2 ks * 1 nt
    __shared__ v16bf sA2[4 * 32];    // stage-2 A fragments (K=128 -> 4 slabs)
    __shared__ v16bf sA3[2 * 32];    // stage-3 A fragments (K=64  -> 2 slabs)
    __shared__ float sB1[128], sB2[64], sB3[16];

    fill_wfrag(sW1, wl1, 64, 128, threadIdx.x, blockDim.x);
    fill_wfrag(sW2, wl2, 128, 64, threadIdx.x, blockDim.x);
    fill_wfrag(sW3, wl3, 64, 16,  threadIdx.x, blockDim.x);
    for (int i = threadIdx.x; i < 128; i += blockDim.x) sB1[i] = bl1[i];
    for (int i = threadIdx.x; i < 64;  i += blockDim.x) sB2[i] = bl2[i];
    for (int i = threadIdx.x; i < 16;  i += blockDim.x) sB3[i] = bl3[i];
    __syncthreads();

    bool compute = threadIdx.x < 32;
    int lane = threadIdx.x & 31;
    int half = lane >> 4;
    int m = lane & 15;
    int n = lane & 15;
    int rowbase = blockIdx.x * 16;
    __bf16* pA2 = (__bf16*)sA2;
    __bf16* pA3 = (__bf16*)sA3;

    // ---- stage 1: [16x64] @ [64x128] -> relu -> sA2 (A-swizzled bf16) ----
    if (compute) {
        v8f acc1[8];
        #pragma unroll
        for (int t = 0; t < 8; ++t) acc1[t] = (v8f){};
        #pragma unroll
        for (int ks = 0; ks < 2; ++ks) {
            v16bf a;
            int row = rowbase + m;
            if (row < G) {
                const float* rp = g + (size_t)row * 64 + ks * 32;
                int k0 = half * 8;
                float4 f0 = *(const float4*)(rp + k0);
                float4 f1 = *(const float4*)(rp + k0 + 4);
                float4 f2 = *(const float4*)(rp + 16 + k0);
                float4 f3 = *(const float4*)(rp + 16 + k0 + 4);
                a[0]=f2bf(f0.x);  a[1]=f2bf(f0.y);  a[2]=f2bf(f0.z);  a[3]=f2bf(f0.w);
                a[4]=f2bf(f1.x);  a[5]=f2bf(f1.y);  a[6]=f2bf(f1.z);  a[7]=f2bf(f1.w);
                a[8]=f2bf(f2.x);  a[9]=f2bf(f2.y);  a[10]=f2bf(f2.z); a[11]=f2bf(f2.w);
                a[12]=f2bf(f3.x); a[13]=f2bf(f3.y); a[14]=f2bf(f3.z); a[15]=f2bf(f3.w);
            } else {
                #pragma unroll
                for (int e = 0; e < 16; ++e) a[e] = f2bf(0.0f);
            }
            #pragma unroll
            for (int nt = 0; nt < 8; ++nt) {
                v16bf b = sW1[(ks * 8 + nt) * 32 + lane];
                acc1[nt] = __builtin_amdgcn_wmma_f32_16x16x32_bf16(
                    false, a, false, b, (short)0, acc1[nt], false, false);
            }
        }
        // epilogue: relu+bias, scatter into stage-2 A-fragment layout
        #pragma unroll
        for (int nt = 0; nt < 8; ++nt) {
            float bv = sB1[nt * 16 + n];
            #pragma unroll
            for (int r = 0; r < 8; ++r) {
                float v = fmaxf(acc1[nt][r] + bv, 0.0f);
                int M = r + 8 * half;
                int N = nt * 16 + n;
                int s = N >> 5, k = N & 31, h, e;
                a_inv(k, h, e);
                pA2[((s * 32) + (M + 16 * h)) * 16 + e] = f2bf(v);
            }
        }
    }
    __syncthreads();

    // ---- stage 2: [16x128] @ [128x64] -> relu -> sA3 ----
    if (compute) {
        v8f acc2[4];
        #pragma unroll
        for (int t = 0; t < 4; ++t) acc2[t] = (v8f){};
        #pragma unroll
        for (int ks = 0; ks < 4; ++ks) {
            v16bf a = sA2[ks * 32 + lane];             // single 32B LDS read
            #pragma unroll
            for (int nt = 0; nt < 4; ++nt) {
                v16bf b = sW2[(ks * 4 + nt) * 32 + lane];
                acc2[nt] = __builtin_amdgcn_wmma_f32_16x16x32_bf16(
                    false, a, false, b, (short)0, acc2[nt], false, false);
            }
        }
        #pragma unroll
        for (int nt = 0; nt < 4; ++nt) {
            float bv = sB2[nt * 16 + n];
            #pragma unroll
            for (int r = 0; r < 8; ++r) {
                float v = fmaxf(acc2[nt][r] + bv, 0.0f);
                int M = r + 8 * half;
                int N = nt * 16 + n;
                int s = N >> 5, k = N & 31, h, e;
                a_inv(k, h, e);
                pA3[((s * 32) + (M + 16 * h)) * 16 + e] = f2bf(v);
            }
        }
    }
    __syncthreads();

    // ---- stage 3: [16x64] @ [64x16] + bias -> out ----
    if (compute) {
        v8f acc3 = (v8f){};
        #pragma unroll
        for (int ks = 0; ks < 2; ++ks) {
            v16bf a = sA3[ks * 32 + lane];
            v16bf b = sW3[ks * 32 + lane];
            acc3 = __builtin_amdgcn_wmma_f32_16x16x32_bf16(
                false, a, false, b, (short)0, acc3, false, false);
        }
        float bv = sB3[n];
        #pragma unroll
        for (int r = 0; r < 8; ++r) {
            int row = rowbase + r + 8 * half;
            if (row < G) out[(size_t)row * 16 + n] = acc3[r] + bv;
        }
    }
}

// ---------- host launcher ----------
extern "C" void kernel_launch(void* const* d_in, const int* in_sizes, int n_in,
                              void* d_out, int out_size, void* d_ws, size_t ws_size,
                              hipStream_t stream) {
    const float* x     = (const float*)d_in[0];
    const int*   ei    = (const int*)  d_in[1];
    const int*   batch = (const int*)  d_in[2];
    // d_in[3] = num_graphs (compile-time constant NG in the reference)
    const float* w1l = (const float*)d_in[4];
    const float* b1l = (const float*)d_in[5];
    const float* w1r = (const float*)d_in[6];
    const float* w2l = (const float*)d_in[7];
    const float* b2l = (const float*)d_in[8];
    const float* w2r = (const float*)d_in[9];
    const float* wl1 = (const float*)d_in[10];
    const float* bl1 = (const float*)d_in[11];
    const float* wl2 = (const float*)d_in[12];
    const float* bl2 = (const float*)d_in[13];
    const float* wl3 = (const float*)d_in[14];
    const float* bl3 = (const float*)d_in[15];
    float* out = (float*)d_out;
    float* ws  = (float*)d_ws;

    // workspace layout: atomically-accumulated buffers first (contiguous zero pass)
    float* agg1 = ws;                         // NN*32
    float* agg2 = agg1 + (size_t)NN * 32;     // NN*64
    float* gbuf = agg2 + (size_t)NN * 64;     // NG*64
    float* h1   = gbuf + (size_t)NG * 64;     // NN*64
    float* h2   = h1   + (size_t)NN * 64;     // NN*64

    size_t nzero = (size_t)NN * 32 + (size_t)NN * 64 + (size_t)NG * 64;
    zero_kernel<<<4096, 256, 0, stream>>>(ws, nzero);

    // layer 1: scatter (wave per edge, lane = feature) then WMMA GEMM
    scatter32_kernel<<<NE * 32 / 256, 256, 0, stream>>>(x, ei, agg1, NE);
    {
        int blocks = (NN / 16 + 7) / 8;            // 8 waves per 256-thread block
        sage_gemm_kernel<32><<<blocks, 256, 0, stream>>>(agg1, x, w1l, w1r, b1l, h1, NN);
    }

    // layer 2
    scatter64_kernel<<<NE * 32 / 256, 256, 0, stream>>>(h1, ei, agg2, NE);
    {
        int blocks = (NN / 16 + 7) / 8;
        sage_gemm_kernel<64><<<blocks, 256, 0, stream>>>(agg2, h1, w2l, w2r, b2l, h2, NN);
    }

    // global add pool
    pool_kernel<<<NN * 32 / 256, 256, 0, stream>>>(h2, batch, gbuf, NN);

    // fused MLP tail: wave0 of each 128-thread block computes 16 graph rows
    {
        int blocks = (NG + 15) / 16;               // 63
        mlp_kernel<<<blocks, 128, 0, stream>>>(gbuf, wl1, bl1, wl2, bl2, wl3, bl3, out, NG);
    }
}
